// RGCN_84490596647379
// MI455X (gfx1250) — matrix operand order repaired and checked by reading, not compiled, not run on hardware
//
#include <hip/hip_runtime.h>

typedef __attribute__((ext_vector_type(16))) _Float16 v16h;
typedef __attribute__((ext_vector_type(8)))  _Float16 v8h;
typedef __attribute__((ext_vector_type(4)))  _Float16 v4h;
typedef __attribute__((ext_vector_type(8)))  float    v8f;

constexpr int NN    = 100000;  // nodes
constexpr int RR    = 4;       // relations
constexpr int EE    = 500000;  // edges per relation
constexpr int IN_F  = 256;
constexpr int HID_F = 128;
constexpr int OUT_F = 64;

// ---------------------------------------------------------------- utilities

__global__ __launch_bounds__(256) void fill_zero(float* p, int n) {
  int i = blockIdx.x * blockDim.x + threadIdx.x;
  if (i < n) p[i] = 0.f;
}

// one thread per (relation, edge): bump out-degree of src and in-degree of dst
__global__ __launch_bounds__(256) void degree_count(
    const int* __restrict__ src, const int* __restrict__ dst,
    float* __restrict__ deg_out, float* __restrict__ deg_in, int total) {
  int i = blockIdx.x * blockDim.x + threadIdx.x;
  if (i >= total) return;
  int r = i / EE;  // compile-time constant divisor
  atomicAdd(&deg_out[(size_t)r * NN + src[i]], 1.0f);
  atomicAdd(&deg_in [(size_t)r * NN + dst[i]], 1.0f);
}

__global__ __launch_bounds__(256) void rnorm_finalize(float* p, int n) {
  int i = blockIdx.x * blockDim.x + threadIdx.x;
  if (i < n) p[i] = rsqrtf(fmaxf(p[i], 1.0f));
}

// out[n][c] = sum_r b[r][c]
template <int C>
__global__ __launch_bounds__(256) void bias_fill(const float* __restrict__ b,
                                                 float* __restrict__ out, int n) {
  int i = blockIdx.x * blockDim.x + threadIdx.x;
  if (i >= n * C) return;
  int c = i & (C - 1);
  out[i] = b[c] + b[C + c] + b[2 * C + c] + b[3 * C + c];
}

// f32 -> f16 (optional fused relu), 4 elements/thread
template <bool RELU>
__global__ __launch_bounds__(256) void cvt_f16(const float* __restrict__ in,
                                               _Float16* __restrict__ out, int n4) {
  int i = blockIdx.x * blockDim.x + threadIdx.x;
  if (i >= n4) return;
  float4 v = ((const float4*)in)[i];
  if (RELU) {
    v.x = fmaxf(v.x, 0.f); v.y = fmaxf(v.y, 0.f);
    v.z = fmaxf(v.z, 0.f); v.w = fmaxf(v.w, 0.f);
  }
  v4h o;
  o[0] = (_Float16)v.x; o[1] = (_Float16)v.y;
  o[2] = (_Float16)v.z; o[3] = (_Float16)v.w;
  ((v4h*)out)[i] = o;
}

// W[r][k][m] (f32) -> Wt[r][m][k] (f16); K*M and M are powers of two
template <int K, int M>
__global__ __launch_bounds__(256) void build_wt(const float* __restrict__ W,
                                                _Float16* __restrict__ Wt) {
  int i = blockIdx.x * blockDim.x + threadIdx.x;
  if (i >= RR * K * M) return;
  int r = i / (K * M);
  int k = (i / M) & (K - 1);
  int m = i & (M - 1);
  Wt[(size_t)r * K * M + (size_t)m * K + k] = (_Float16)W[i];
}

// ---------------------------------------------------------------- WMMA GEMM
// Ch[n x M] (f16) = (A[n x K] (f16) @ Wt^T) * scale[n]
// Wt is [M][K] f16. Block: 256 threads = 8 wave32; each wave owns a 16-row strip.
// A fragments load straight from global (ISA 16x32 f16 A layout = two 16B runs/lane);
// whole Wt staged to LDS once per block (single barrier), B frags via ds_load_b128.
template <int K, int M>
__global__ __launch_bounds__(256) void gemm_f16(
    const _Float16* __restrict__ A, const _Float16* __restrict__ Wt,
    const float* __restrict__ scale, _Float16* __restrict__ Ch, int n) {
  constexpr int WSTR = K + 8;  // padded row stride (16B-aligned, conflict-spread)
  __shared__ _Float16 Wlds[M * WSTR];

  const int t    = threadIdx.x;
  const int lane = t & 31;
  const int w    = t >> 5;
  const int half = lane >> 4;
  const int l16  = lane & 15;

  // ---- stage whole Wt into LDS (8-byte chunks, coalesced)
  const unsigned long long* ws = (const unsigned long long*)Wt;
#pragma unroll
  for (int i = t; i < K * M / 4; i += 256) {
    int m  = i / (K / 4);
    int kq = i - m * (K / 4);
    *(unsigned long long*)&Wlds[m * WSTR + kq * 4] = ws[i];
  }
  __syncthreads();

  // ---- this lane's A row (clamped; OOB rows only feed OOB C rows, never stored)
  int arow = blockIdx.x * 128 + w * 16 + l16;
  const _Float16* ap = A + (size_t)(arow < n ? arow : n - 1) * K + half * 8;

  v8f acc[M / 16];
#pragma unroll
  for (int ct = 0; ct < M / 16; ++ct) acc[ct] = v8f{};

  for (int k0 = 0; k0 < K; k0 += 32) {
    // A fragment: K = half*8 + {0..7} and half*8 + 16 + {0..7}
    v8h alo = *(const v8h*)(ap + k0);
    v8h ahi = *(const v8h*)(ap + k0 + 16);
    v16h a;
#pragma unroll
    for (int i = 0; i < 8; ++i) { a[i] = alo[i]; a[8 + i] = ahi[i]; }

#pragma unroll
    for (int ct = 0; ct < M / 16; ++ct) {
      // B fragment: col N = l16, K = half*8 + {0..7} and half*8 + 16 + {0..7}
      const _Float16* bp = &Wlds[(ct * 16 + l16) * WSTR + k0 + half * 8];
      v8h blo = *(const v8h*)bp;
      v8h bhi = *(const v8h*)(bp + 16);
      v16h b;
#pragma unroll
      for (int i = 0; i < 8; ++i) { b[i] = blo[i]; b[8 + i] = bhi[i]; }
      acc[ct] = __builtin_amdgcn_wmma_f32_16x16x32_f16(
          false, a, false, b, (short)0, acc[ct], false, false);
    }
  }

  // ---- scale rows ( (s*x)@W == s*(x@W) ) and store f16
  const int rbase = blockIdx.x * 128 + w * 16 + half * 8;
  float sv[8];
#pragma unroll
  for (int v = 0; v < 8; ++v) {
    int g = rbase + v;
    sv[v] = (g < n) ? scale[g] : 0.f;
  }
#pragma unroll
  for (int ct = 0; ct < M / 16; ++ct) {
#pragma unroll
    for (int v = 0; v < 8; ++v) {
      int g = rbase + v;
      if (g < n) Ch[(size_t)g * M + ct * 16 + l16] = (_Float16)(acc[ct][v] * sv[v]);
    }
  }
}

// ---------------------------------------------------------------- scatter-add
// out[dst[e]][:] += f32(h[src[e]][:]) * rnorm_in[dst[e]]   (f16 gather, f32 atomics)
// lane = 8-col chunk -> 16B b128 gathers; src/dst loads uniform per lane group
template <int C>
__global__ __launch_bounds__(256) void scatter_add(
    const _Float16* __restrict__ h, const int* __restrict__ src,
    const int* __restrict__ dst, const float* __restrict__ rin,
    float* __restrict__ out, int e_cnt) {
  constexpr int CH = C / 8;
  int idx = blockIdx.x * blockDim.x + threadIdx.x;
  int e = idx / CH;
  int c = idx - e * CH;
  if (e >= e_cnt) return;
  int s = src[e];
  int d = dst[e];
  float sc = rin[d];
  v8h v = *(const v8h*)&h[(size_t)s * C + c * 8];
  float* o = &out[(size_t)d * C + c * 8];
#pragma unroll
  for (int i = 0; i < 8; ++i) atomicAdd(o + i, (float)v[i] * sc);
}

// ---------------------------------------------------------------- launcher

static inline int cdiv(long a, long b) { return (int)((a + b - 1) / b); }

extern "C" void kernel_launch(void* const* d_in, const int* in_sizes, int n_in,
                              void* d_out, int out_size, void* d_ws, size_t ws_size,
                              hipStream_t stream) {
  const float* x    = (const float*)d_in[0];
  const int*   esrc = (const int*)d_in[1];
  const int*   edst = (const int*)d_in[2];
  const float* W1   = (const float*)d_in[3];
  const float* b1   = (const float*)d_in[4];
  const float* W2   = (const float*)d_in[5];
  const float* b2   = (const float*)d_in[6];
  float* out = (float*)d_out;

  // workspace (~132 MB):
  //   rno[R*N] f32 | rni[R*N] f32 | h1[N*HID] f32 |
  //   hbuf[N*HID] f16 | xh[N*IN] f16 (reused as h1h[N*HID] f16) | wt1 | wt2
  float*    rno  = (float*)d_ws;
  float*    rni  = rno + (size_t)RR * NN;
  float*    h1   = rni + (size_t)RR * NN;
  _Float16* hbuf = (_Float16*)(h1 + (size_t)NN * HID_F);
  _Float16* xh   = hbuf + (size_t)NN * HID_F;
  _Float16* wt1  = xh + (size_t)NN * IN_F;
  _Float16* wt2  = wt1 + (size_t)RR * IN_F * HID_F;
  _Float16* h1h  = xh;  // layer-2 input aliases xh (x no longer needed)

  // degrees -> rsqrt(max(deg,1)), shared by both layers
  fill_zero<<<cdiv(2L * RR * NN, 256), 256, 0, stream>>>(rno, 2 * RR * NN);
  degree_count<<<cdiv((long)RR * EE, 256), 256, 0, stream>>>(esrc, edst, rno, rni,
                                                             RR * EE);
  rnorm_finalize<<<cdiv(2L * RR * NN, 256), 256, 0, stream>>>(rno, 2 * RR * NN);

  // one-time f16 conversions
  cvt_f16<false><<<cdiv((long)NN * IN_F / 4, 256), 256, 0, stream>>>(
      x, xh, NN * IN_F / 4);
  build_wt<IN_F, HID_F><<<cdiv((long)RR * IN_F * HID_F, 256), 256, 0, stream>>>(W1, wt1);
  build_wt<HID_F, OUT_F><<<cdiv((long)RR * HID_F * OUT_F, 256), 256, 0, stream>>>(W2, wt2);

  // ---- layer 1: h1 = sum_r scatter_r( (xh @ W1_r) * rno_r ) * rni_r + sum_r b1_r
  bias_fill<HID_F><<<cdiv((long)NN * HID_F, 256), 256, 0, stream>>>(b1, h1, NN);
  for (int r = 0; r < RR; ++r) {
    gemm_f16<IN_F, HID_F><<<cdiv(NN, 128), 256, 0, stream>>>(
        xh, wt1 + (size_t)r * IN_F * HID_F, rno + (size_t)r * NN, hbuf, NN);
    scatter_add<HID_F><<<cdiv((long)EE * (HID_F / 8), 256), 256, 0, stream>>>(
        hbuf, esrc + (size_t)r * EE, edst + (size_t)r * EE,
        rni + (size_t)r * NN, h1, EE);
  }

  // relu(h1) -> f16 once (layer-2 GEMM input)
  cvt_f16<true><<<cdiv((long)NN * HID_F / 4, 256), 256, 0, stream>>>(
      h1, h1h, NN * HID_F / 4);

  // ---- layer 2
  bias_fill<OUT_F><<<cdiv((long)NN * OUT_F, 256), 256, 0, stream>>>(b2, out, NN);
  for (int r = 0; r < RR; ++r) {
    gemm_f16<HID_F, OUT_F><<<cdiv(NN, 128), 256, 0, stream>>>(
        h1h, wt2 + (size_t)r * HID_F * OUT_F, rno + (size_t)r * NN, hbuf, NN);
    scatter_add<OUT_F><<<cdiv((long)EE * (OUT_F / 8), 256), 256, 0, stream>>>(
        hbuf, esrc + (size_t)r * EE, edst + (size_t)r * EE,
        rni + (size_t)r * NN, out, EE);
  }
}
